// Net_23905787969856
// MI455X (gfx1250) — matrix operand order repaired and checked by reading, not compiled, not run on hardware
//
#include <hip/hip_runtime.h>

typedef _Float16 f16;
typedef _Float16 v16h __attribute__((ext_vector_type(16)));
typedef _Float16 v8h  __attribute__((ext_vector_type(8)));
typedef float    v8f  __attribute__((ext_vector_type(8)));

union AV { v8h h[2]; v16h v; };

#define BATCH 4096

// ---------------------------------------------------------------------------
// Repack W2 [64,32,3,3] f32 (OIHW) -> Bw [N=64][K=288] f16, K = (3*ky+kx)*32 + ic
// ---------------------------------------------------------------------------
__global__ __launch_bounds__(256) void repack_w2(const float* __restrict__ W2,
                                                 f16* __restrict__ Bw) {
    int idx = blockIdx.x * 256 + threadIdx.x;      // idx = oc*288 + k
    if (idx >= 288 * 64) return;
    int oc = idx / 288, k = idx % 288;
    int t = k >> 5, c = k & 31;
    int ky = t / 3, kx = t % 3;
    Bw[idx] = (f16)W2[((oc * 32 + c) * 3 + ky) * 3 + kx];
}

// ---------------------------------------------------------------------------
// Repack Wl1 [12544][128] f32 (rows in NCHW flatten order oc*196+y*14+x)
// -> Wp [128][12544] f16 with k in our NHWC order r = y*896 + x*64 + oc
// ---------------------------------------------------------------------------
__global__ __launch_bounds__(256) void repack_wl1(const float* __restrict__ Wl1,
                                                  f16* __restrict__ Wp) {
    int idx = blockIdx.x * 256 + threadIdx.x;      // idx = j*12544 + r
    int j = idx / 12544, r = idx % 12544;
    int y = r / 896, rem = r % 896;
    int xp = rem >> 6, oc = rem & 63;
    Wp[idx] = (f16)Wl1[(size_t)(oc * 196 + y * 14 + xp) * 128 + j];
}

// ---------------------------------------------------------------------------
// conv1: 1->32 ch, 3x3 SAME, +bias, *mask, ReLU. Output NHWC f16 [B,28,28,32].
// One thread per pixel, all 32 channels (288 FMA) -> ~1.9 GFLOP, VALU-bound.
// ---------------------------------------------------------------------------
__global__ __launch_bounds__(256) void conv1_kernel(const float* __restrict__ x,
                                                    const float* __restrict__ W1,
                                                    const float* __restrict__ b1,
                                                    f16* __restrict__ h1) {
    int idx = blockIdx.x * 256 + threadIdx.x;      // over B*784
    if (idx >= BATCH * 784) return;
    int b = idx / 784, p = idx % 784;
    int y = p / 28, xx = p % 28;
    const float* xb = x + (size_t)b * 784;
    float center = xb[y * 28 + xx];
    bool active = (center != 0.0f);
    float in[9];
#pragma unroll
    for (int t = 0; t < 9; ++t) {
        int iy = y + t / 3 - 1, ix = xx + t % 3 - 1;
        in[t] = (iy >= 0 && iy < 28 && ix >= 0 && ix < 28) ? xb[iy * 28 + ix] : 0.0f;
    }
    f16* outp = h1 + (size_t)idx * 32;
#pragma unroll
    for (int c4 = 0; c4 < 4; ++c4) {
        v8h pk;
#pragma unroll
        for (int e = 0; e < 8; ++e) {
            int oc = c4 * 8 + e;
            float acc = b1[oc];
#pragma unroll
            for (int t = 0; t < 9; ++t) acc = fmaf(W1[oc * 9 + t], in[t], acc);
            float v = active ? fmaxf(acc, 0.0f) : 0.0f;
            pk[e] = (f16)v;
        }
        *(v8h*)(outp + c4 * 8) = pk;     // 16B store
    }
}

// ---------------------------------------------------------------------------
// conv2 (32->64, 3x3 SAME) + bias + mask + ReLU + fused 2x2/2 max-pool.
// Implicit GEMM: per block (b, pooled row r): M=64 pixels (2 rows x 28 cols,
// 8 dummies), N=64 out-ch, K=288 (9 taps x 32 ch). 4 waves, each one M-tile
// x 4 N-tiles; 9 K-steps of v_wmma_f32_16x16x32_f16. Pool via LDS epilogue.
// Output h2 NHWC f16 [B,14,14,64].
// ---------------------------------------------------------------------------
__global__ __launch_bounds__(128) void conv2_pool_kernel(const f16* __restrict__ h1,
                                                         const f16* __restrict__ Bw,
                                                         const float* __restrict__ b2,
                                                         const float* __restrict__ x,
                                                         f16* __restrict__ h2) {
    __shared__ float lds[56 * 65];                 // stride 65: no 2-way bank conflict
    int blk = blockIdx.x;
    int b = blk / 14, r = blk % 14;
    int y0 = 2 * r;
    int tid = threadIdx.x;
    int w = tid >> 5, lane = tid & 31;
    int m16 = lane & 15, hi = lane >> 4;

    // A-layout: this lane supplies row m = m16; chunks at k = 8*hi+{0..7}, +16..+23
    int p = w * 16 + m16;
    int pc = (p < 56) ? p : 55;                    // clamp dummy pixels
    int ry = pc / 28, rx = pc % 28;
    int yy = y0 + ry;

    v8f acc[4] = {};
#pragma unroll
    for (int t = 0; t < 9; ++t) {
        int iy = yy + t / 3 - 1, ix = rx + t % 3 - 1;
        v8h lo = {}, hi8 = {};
        if (iy >= 0 && iy < 28 && ix >= 0 && ix < 28) {
            const f16* src = h1 + ((size_t)(b * 28 + iy) * 28 + ix) * 32 + 8 * hi;
            lo  = *(const v8h*)(src);              // channels kb..kb+7
            hi8 = *(const v8h*)(src + 16);         // channels kb+16..kb+23
        }
        AV ua; ua.h[0] = lo; ua.h[1] = hi8;
        v16h a = ua.v;
        // B-layout: this lane supplies col n = j*16 + m16; k = 8*hi+{0..7}, +16..+23
        const f16* brow = Bw + (size_t)m16 * 288 + t * 32 + 8 * hi;
#pragma unroll
        for (int j = 0; j < 4; ++j) {
            const f16* bp = brow + j * 16 * 288;
            AV ub; ub.h[0] = *(const v8h*)(bp); ub.h[1] = *(const v8h*)(bp + 16);
            v16h bm = ub.v;
            acc[j] = __builtin_amdgcn_wmma_f32_16x16x32_f16(
                false, a, false, bm, (short)0, acc[j], false, false);
        }
    }

    // epilogue: bias + sparsity mask + ReLU -> LDS
    const float* xb = x + (size_t)b * 784;
#pragma unroll
    for (int rr = 0; rr < 8; ++rr) {
        int pp = w * 16 + rr + 8 * hi;             // C row M = rr + 8*hi
        if (pp < 56) {
            int py = y0 + pp / 28, px = pp % 28;
            bool active = (xb[py * 28 + px] != 0.0f);
#pragma unroll
            for (int j = 0; j < 4; ++j) {
                int oc = j * 16 + m16;             // C col N = m16
                float v = acc[j][rr] + b2[oc];
                v = active ? fmaxf(v, 0.0f) : 0.0f;
                lds[pp * 65 + oc] = v;
            }
        }
    }
    __syncthreads();

    // fused 2x2 max-pool: 14*64 = 896 outputs
    for (int i = tid; i < 14 * 64; i += 128) {
        int px = i >> 6, oc = i & 63;
        float v00 = lds[(2 * px)      * 65 + oc];
        float v01 = lds[(2 * px + 1)  * 65 + oc];
        float v10 = lds[(28 + 2 * px) * 65 + oc];
        float v11 = lds[(28 + 2 * px + 1) * 65 + oc];
        float v = fmaxf(fmaxf(v00, v01), fmaxf(v10, v11));
        h2[(((size_t)b * 14 + r) * 14 + px) * 64 + oc] = (f16)v;
    }
}

// ---------------------------------------------------------------------------
// FC1: out[4096][128] = relu(h2flat[4096][12544] @ Wp[12544][128] + bias).
// 64 blocks x 4 waves; wave = one M-tile (16 rows) x 8 N-tiles, 392 K-steps.
// ---------------------------------------------------------------------------
__global__ __launch_bounds__(128) void fc1_kernel(const f16* __restrict__ A,
                                                  const f16* __restrict__ Bw,
                                                  const float* __restrict__ bias,
                                                  float* __restrict__ out) {
    int tid = threadIdx.x;
    int w = tid >> 5, lane = tid & 31;
    int m16 = lane & 15, hi = lane >> 4;
    int base = (blockIdx.x * 4 + w) * 16;
    const f16* arow  = A  + (size_t)(base + m16) * 12544 + 8 * hi;
    const f16* bbase = Bw + (size_t)m16 * 12544 + 8 * hi;

    v8f acc[8] = {};
    for (int ks = 0; ks < 392; ++ks) {
        __builtin_prefetch(arow + (ks + 8) * 32, 0, 1);   // global_prefetch_b8
        AV ua;
        ua.h[0] = *(const v8h*)(arow + ks * 32);
        ua.h[1] = *(const v8h*)(arow + ks * 32 + 16);
        v16h a = ua.v;
        const f16* brow = bbase + (size_t)ks * 32;
#pragma unroll
        for (int j = 0; j < 8; ++j) {
            const f16* bp = brow + (size_t)j * 16 * 12544;
            AV ub; ub.h[0] = *(const v8h*)(bp); ub.h[1] = *(const v8h*)(bp + 16);
            v16h bm = ub.v;
            acc[j] = __builtin_amdgcn_wmma_f32_16x16x32_f16(
                false, a, false, bm, (short)0, acc[j], false, false);
        }
    }
#pragma unroll
    for (int j = 0; j < 8; ++j) {
#pragma unroll
        for (int rr = 0; rr < 8; ++rr) {
            int m = base + rr + 8 * hi;
            int col = j * 16 + m16;
            float v = acc[j][rr] + bias[col];
            out[(size_t)m * 128 + col] = fmaxf(v, 0.0f);
        }
    }
}

// ---------------------------------------------------------------------------
// FC2 (128->10) + log_softmax. One wave32 per batch row; __shfl_xor reduce.
// ---------------------------------------------------------------------------
__global__ __launch_bounds__(32) void fc2_lsm_kernel(const float* __restrict__ h,
                                                     const float* __restrict__ W,
                                                     const float* __restrict__ bias,
                                                     float* __restrict__ out) {
    int b = blockIdx.x, lane = threadIdx.x;
    float hv[4];
#pragma unroll
    for (int i = 0; i < 4; ++i) hv[i] = h[(size_t)b * 128 + lane + 32 * i];
    float logits[10];
#pragma unroll
    for (int c = 0; c < 10; ++c) {
        float s = 0.0f;
#pragma unroll
        for (int i = 0; i < 4; ++i)
            s = fmaf(hv[i], W[(size_t)(lane + 32 * i) * 10 + c], s);
#pragma unroll
        for (int off = 16; off >= 1; off >>= 1) s += __shfl_xor(s, off, 32);
        logits[c] = s + bias[c];                 // full sum in every lane
    }
    float mx = logits[0];
#pragma unroll
    for (int c = 1; c < 10; ++c) mx = fmaxf(mx, logits[c]);
    float se = 0.0f;
#pragma unroll
    for (int c = 0; c < 10; ++c) se += expf(logits[c] - mx);
    float lse = mx + logf(se);
    float mine = 0.0f;
#pragma unroll
    for (int c = 0; c < 10; ++c)
        if (lane == c) mine = logits[c];
    if (lane < 10) out[(size_t)b * 10 + lane] = mine - lse;
}

// ---------------------------------------------------------------------------
extern "C" void kernel_launch(void* const* d_in, const int* in_sizes, int n_in,
                              void* d_out, int out_size, void* d_ws, size_t ws_size,
                              hipStream_t stream) {
    (void)in_sizes; (void)n_in; (void)out_size; (void)ws_size;
    const float* x   = (const float*)d_in[0];
    const float* W1  = (const float*)d_in[1];
    const float* b1  = (const float*)d_in[2];
    const float* W2  = (const float*)d_in[3];
    const float* b2  = (const float*)d_in[4];
    const float* Wl1 = (const float*)d_in[5];
    const float* bl1 = (const float*)d_in[6];
    const float* Wl2 = (const float*)d_in[7];
    const float* bl2 = (const float*)d_in[8];
    float* out = (float*)d_out;

    char* ws = (char*)d_ws;
    f16*   h1   = (f16*)(ws);                      // [B,28,28,32] f16 : 205,520,896 B
    f16*   h2   = (f16*)(ws + 205520896);          // [B,14,14,64] f16 : 102,760,448 B
    float* f1   = (float*)(ws + 308281344);        // [B,128] f32      :   2,097,152 B
    f16*   Bw2  = (f16*)(ws + 310378496);          // [288,64] f16     :      36,864 B
    f16*   Wl1p = (f16*)(ws + 310415360);          // [12544,128] f16  :   3,211,264 B
                                                   // total ~313.6 MB of d_ws

    hipLaunchKernelGGL(repack_w2,  dim3(72),   dim3(256), 0, stream, W2, Bw2);
    hipLaunchKernelGGL(repack_wl1, dim3(6272), dim3(256), 0, stream, Wl1, Wl1p);
    hipLaunchKernelGGL(conv1_kernel, dim3((BATCH * 784) / 256), dim3(256), 0, stream,
                       x, W1, b1, h1);
    hipLaunchKernelGGL(conv2_pool_kernel, dim3(BATCH * 14), dim3(128), 0, stream,
                       h1, Bw2, b2, x, h2);
    hipLaunchKernelGGL(fc1_kernel, dim3(64), dim3(128), 0, stream, h2, Wl1p, bl1, f1);
    hipLaunchKernelGGL(fc2_lsm_kernel, dim3(BATCH), dim3(32), 0, stream, f1, Wl2, bl2, out);
}
